// Hybrid_13022340842114
// MI455X (gfx1250) — matrix operand order repaired and checked by reading, not compiled, not run on hardware
//
#include <hip/hip_runtime.h>
#include <stdint.h>

// Problem dims (match reference)
#define SEQ_LEN 8192
#define NS      256        // states
#define NI      512        // inputs
#define NO      512        // outputs
#define NC      2          // counters
#define KDIM    258        // NS + NC
#define KPAD    288        // padded to 9 * 32 for WMMA bf16 K-steps

// LDS slice geometry in the scan kernel: TDM pads each 512 B row by 16 B.
#define ROWB    528        // bytes per padded row in LDS
#define SLICEB  135168     // 256 * 528
#define ST_OFF  270336     // 2 * SLICEB
#define CNT_OFF 270848     // ST_OFF + 512
#define SMEMB   270912

typedef __attribute__((ext_vector_type(16))) __bf16 v16bf;
typedef __attribute__((ext_vector_type(8)))  __bf16 v8bf;
typedef __attribute__((ext_vector_type(8)))  float  v8f;
typedef __attribute__((ext_vector_type(4)))  unsigned int u32x4;
typedef __attribute__((ext_vector_type(8)))  int i32x8;
typedef __attribute__((ext_vector_type(4)))  int i32x4;

// ---------------------------------------------------------------------------
// Kernel 1: row softmax of T_param [S, I, S] -> bf16 Tsm TRANSPOSED per
// symbol: Tsm[inp][j][i] = softmax_j(T_param[i, inp, :])[j].  Transposed so
// the scan kernel's WMMA B-fragments read K(=i) contiguously from LDS.
// Block: 256 threads = 8 waves; wave w softmaxes source row i0+w of symbol
// inp, then the block transposes 8x256 through LDS and stores 16B chunks.
// ---------------------------------------------------------------------------
__global__ __launch_bounds__(256) void softmax_T_kernel(
    const float* __restrict__ Tp, __bf16* __restrict__ Tsm) {
  __shared__ __bf16 sb[8 * NS];                     // 4 KB transpose staging
  const int inp  = blockIdx.x & (NI - 1);
  const int i0   = (blockIdx.x >> 9) * 8;
  const int lane = threadIdx.x & 31;
  const int wave = threadIdx.x >> 5;
  const int i    = i0 + wave;

  const float* row = Tp + ((size_t)i * NI + inp) * NS + lane * 8;
  float4 a = *(const float4*)row;
  float4 b = *(const float4*)(row + 4);
  float x[8] = {a.x, a.y, a.z, a.w, b.x, b.y, b.z, b.w};
  float mx = x[0];
#pragma unroll
  for (int k = 1; k < 8; ++k) mx = fmaxf(mx, x[k]);
#pragma unroll
  for (int msk = 16; msk >= 1; msk >>= 1) mx = fmaxf(mx, __shfl_xor(mx, msk, 32));
  float e[8];
  float s = 0.f;
#pragma unroll
  for (int k = 0; k < 8; ++k) { e[k] = __expf(x[k] - mx); s += e[k]; }
#pragma unroll
  for (int msk = 16; msk >= 1; msk >>= 1) s += __shfl_xor(s, msk, 32);
  const float inv = 1.f / s;
#pragma unroll
  for (int k = 0; k < 8; ++k) sb[wave * NS + lane * 8 + k] = (__bf16)(e[k] * inv);
  __syncthreads();

  const int j = threadIdx.x;                        // destination row (state j)
  v8bf o;
#pragma unroll
  for (int w = 0; w < 8; ++w) o[w] = sb[w * NS + j];
  *(v8bf*)(Tsm + ((size_t)inp * NS + j) * NS + i0) = o;
}

// ---------------------------------------------------------------------------
// Kernel 2: pack out_W [O, 258] f32 -> bf16 [O, 288] (zero K-padding).
// ---------------------------------------------------------------------------
__global__ __launch_bounds__(256) void pack_w_kernel(
    const float* __restrict__ W, __bf16* __restrict__ Wb) {
  int idx = blockIdx.x * 256 + threadIdx.x;         // < NO * KPAD
  int o = idx / KPAD;
  int k = idx - o * KPAD;
  float v = (k < KDIM) ? W[o * KDIM + k] : 0.f;
  Wb[idx] = (__bf16)v;
}

// ---------------------------------------------------------------------------
// TDM: async 2D tile load (256 x 256 bf16 slice) global -> LDS, with LDS
// destination padding: 4 dwords (16 B) after every 128 dwords (one 512 B
// row) -> padded LDS row stride 528 B (keeps b128 alignment, spreads banks).
// D# packed per CDNA5 ISA ch.8.
// ---------------------------------------------------------------------------
__device__ __forceinline__ void tdm_load_slice(unsigned int lds_off,
                                               const void* gptr) {
  uint64_t ga = (uint64_t)(uintptr_t)gptr;
  u32x4 g0;
  g0[0] = 1u;                                         // count=1, user mode
  g0[1] = lds_off;                                    // LDS byte address
  g0[2] = (unsigned int)ga;                           // global addr [31:0]
  g0[3] = ((unsigned int)(ga >> 32) & 0x01FFFFFFu)    // global addr [56:32]
          | 0x80000000u;                              // type = 2 ("image")
  i32x8 g1;
  g1[0] = 0x00010000                                  // data_size=1 (2 bytes)
        | (1 << 20)                                   // pad_enable
        | (6 << 22)                                   // pad_interval: 128 dw
        | (3 << 25);                                  // pad_amount: 4 dwords
  g1[1] = 0x01000000;   // tensor_dim0 low16 = 256 (bits 63:48)
  g1[2] = 0x01000000;   // tensor_dim0 hi=0, tensor_dim1 low16 = 256
  g1[3] = 0x01000000;   // tensor_dim1 hi=0, tile_dim0 = 256
  g1[4] = 0x00000100;   // tile_dim1 = 256, tile_dim2 = 0
  g1[5] = 0x00000100;   // tensor_dim0_stride = 256 (low 32)
  g1[6] = 0;
  g1[7] = 0;
  i32x4 z4 = {0, 0, 0, 0};
#if defined(__clang_major__) && (__clang_major__ >= 23)
  i32x8 z8 = {0, 0, 0, 0, 0, 0, 0, 0};
  __builtin_amdgcn_tensor_load_to_lds(g0, g1, z4, z4, z8, 0);
#else
  __builtin_amdgcn_tensor_load_to_lds(g0, g1, z4, z4, 0);
#endif
}

// ---------------------------------------------------------------------------
// Kernel 3: sequential scan, WMMA edition. Single workgroup, 256 thr, 8 waves.
// LDS: [0,132K) slice buf A, [132K,264K) buf B (TDM double buffer, padded
// rows), then bf16 state[256] and f32 counters[2].
//
// new_state(1x256) = state(1x256) @ M(256x256): the A-fragment carries the
// state vector broadcast into ALL 16 rows (uniform LDS loads, contiguous
// k = kb + 16*half + s, as B), so every row of D equals the new state segment and each
// lane reads its own column's result from acc[0] — no extraction shuffle.
// B-fragments come straight out of the TDM-transposed padded slice.
// Wave w owns output columns [32w, 32w+32) = two 16-wide N-tiles; 8 K-steps.
// Wave 0 TDM-prefetches step t+1's slice while all waves compute step t.
// ---------------------------------------------------------------------------
__global__ __launch_bounds__(256) void scan_kernel(
    const int* __restrict__ seq, const __bf16* __restrict__ Tsm,
    const float* __restrict__ delta, __bf16* __restrict__ feats) {
  extern __shared__ char smem[];
  __bf16* st_bf = (__bf16*)(smem + ST_OFF);
  float*  cnt   = (float*)(smem + CNT_OFF);
  const int tid  = threadIdx.x;
  const int wave = tid >> 5;
  const int lane = tid & 31;
  const int m    = lane & 15;
  const int half = lane >> 4;
  const int j0   = wave * 32;                       // wave's 32 output columns

  st_bf[tid] = (__bf16)((tid == 0) ? 1.f : 0.f);    // state0 = e0
  if (tid < NC) cnt[tid] = 0.f;

  if (tid == 0) {                                   // preload slice for t=0
    tdm_load_slice(0u, Tsm + (size_t)seq[0] * (NS * NS));
  }
  __builtin_amdgcn_s_wait_tensorcnt(0);
  __syncthreads();

  for (int t = 0; t < SEQ_LEN; ++t) {
    const char* mb = smem + ((t & 1) ? SLICEB : 0);
    if (tid == 0 && (t + 1) < SEQ_LEN) {            // async-prefetch t+1 slice
      tdm_load_slice(((t + 1) & 1) ? (unsigned)SLICEB : 0u,
                     Tsm + (size_t)seq[t + 1] * (NS * NS));
    }
    const int inp = seq[t];
    if (tid < NC) cnt[tid] += delta[tid * NI + inp];

    v8f acc0 = v8f{};
    v8f acc1 = v8f{};
#pragma unroll
    for (int kb = 0; kb < NS; kb += 32) {
      // A 16x32: st[k] replicated into every row (broadcast LDS loads;
      // half 0 covers K=kb+{0..15}, half 1 covers +16).
      v8bf zlo = *(const v8bf*)(st_bf + kb + half * 16);
      v8bf zhi = *(const v8bf*)(st_bf + kb + half * 16 + 8);
      v16bf A;
#pragma unroll
      for (int e = 0; e < 8; ++e) { A[e] = zlo[e]; A[e + 8] = zhi[e]; }
      // B 32x16: N = lane&15 -> row j of transposed slice, K contiguous.
      const char* r0 = mb + (size_t)(j0 + m) * ROWB + 2 * kb + 32 * half;
      const char* r1 = mb + (size_t)(j0 + 16 + m) * ROWB + 2 * kb + 32 * half;
      v8bf b0l = *(const v8bf*)r0;
      v8bf b0h = *(const v8bf*)(r0 + 16);
      v8bf b1l = *(const v8bf*)r1;
      v8bf b1h = *(const v8bf*)(r1 + 16);
      v16bf B0, B1;
#pragma unroll
      for (int e = 0; e < 8; ++e) {
        B0[e] = b0l[e]; B0[e + 8] = b0h[e];
        B1[e] = b1l[e]; B1[e + 8] = b1h[e];
      }
      acc0 = __builtin_amdgcn_wmma_f32_16x16x32_bf16(
          false, A, false, B0, (short)0, acc0, false, false);
      acc1 = __builtin_amdgcn_wmma_f32_16x16x32_bf16(
          false, A, false, B1, (short)0, acc1, false, false);
    }
    __syncthreads();                                // A reads done before write

    // Every D row is identical; element 0 of acc = new_state[column].
    const float v0 = acc0[0];
    const float v1 = acc1[0];
    __bf16* frow = feats + (size_t)t * KPAD;
    if (lane < 16) {                                // halves are duplicates
      const int jA = j0 + m;
      const int jB = j0 + 16 + m;
      st_bf[jA] = (__bf16)v0;
      st_bf[jB] = (__bf16)v1;
      frow[jA] = (__bf16)v0;
      frow[jB] = (__bf16)v1;
    }
    if (tid < NC) frow[NS + tid] = (__bf16)cnt[tid];
    else if (tid < 32) frow[NS + tid] = (__bf16)0.f; // zero pad 258..287

    __builtin_amdgcn_s_wait_tensorcnt(0);           // t+1 slice landed
    __syncthreads();                                // publish LDS to all waves
  }
}

// ---------------------------------------------------------------------------
// Kernel 4: output GEMM + bias + row softmax (unchanged; compiles to clean
// pipelined v_wmma_f32_16x16x32_bf16 + clustered global_load_b128).
// ---------------------------------------------------------------------------
__global__ __launch_bounds__(256) void out_gemm_kernel(
    const __bf16* __restrict__ feats, const __bf16* __restrict__ Wb,
    const float* __restrict__ bias, float* __restrict__ out) {
  __shared__ float sl[16 * NO];                     // 32 KB logits tile

  const int t0   = blockIdx.x * 16;
  const int wave = threadIdx.x >> 5;
  const int lane = threadIdx.x & 31;
  const int m    = lane & 15;
  const int half = lane >> 4;

  const __bf16* arow = feats + (size_t)(t0 + m) * KPAD + half * 16;
  v8f acc[4] = {v8f{}, v8f{}, v8f{}, v8f{}};

  for (int kb = 0; kb < KPAD; kb += 32) {
    v8bf alo = *(const v8bf*)(arow + kb);
    v8bf ahi = *(const v8bf*)(arow + kb + 8);
    v16bf A;
#pragma unroll
    for (int e = 0; e < 8; ++e) { A[e] = alo[e]; A[e + 8] = ahi[e]; }
#pragma unroll
    for (int n = 0; n < 4; ++n) {
      const int ocol = (wave * 4 + n) * 16 + m;
      v16bf B = *(const v16bf*)(Wb + (size_t)ocol * KPAD + kb + half * 16);
      acc[n] = __builtin_amdgcn_wmma_f32_16x16x32_bf16(
          false, A, false, B, (short)0, acc[n], false, false);
    }
  }

#pragma unroll
  for (int n = 0; n < 4; ++n) {
    const int ocol = (wave * 4 + n) * 16 + m;
    const float b = bias[ocol];
#pragma unroll
    for (int r = 0; r < 8; ++r) {
      sl[(r + 8 * half) * NO + ocol] = acc[n][r] + b;
    }
  }
  __syncthreads();

  const int rr = threadIdx.x >> 4;
  const int c0 = threadIdx.x & 15;
  float mx = -3.402823466e+38f;
  for (int c = c0; c < NO; c += 16) mx = fmaxf(mx, sl[rr * NO + c]);
#pragma unroll
  for (int msk = 8; msk >= 1; msk >>= 1) mx = fmaxf(mx, __shfl_xor(mx, msk, 32));
  float s = 0.f;
  for (int c = c0; c < NO; c += 16) s += __expf(sl[rr * NO + c] - mx);
#pragma unroll
  for (int msk = 8; msk >= 1; msk >>= 1) s += __shfl_xor(s, msk, 32);
  const float inv = 1.f / s;
  float* orow = out + (size_t)(t0 + rr) * NO;
  for (int c = c0; c < NO; c += 16) orow[c] = __expf(sl[rr * NO + c] - mx) * inv;
}

// ---------------------------------------------------------------------------
// Host launcher
// ---------------------------------------------------------------------------
extern "C" void kernel_launch(void* const* d_in, const int* in_sizes, int n_in,
                              void* d_out, int out_size, void* d_ws,
                              size_t ws_size, hipStream_t stream) {
  const int*   seq   = (const int*)d_in[0];
  const float* Tp    = (const float*)d_in[1];
  const float* delta = (const float*)d_in[2];
  const float* W     = (const float*)d_in[3];
  const float* bias  = (const float*)d_in[4];
  float*       out   = (float*)d_out;

  char* ws = (char*)d_ws;
  __bf16* Tsm   = (__bf16*)ws;                         // 67,108,864 B
  __bf16* feats = (__bf16*)(ws + 67108864);            //  4,718,592 B
  __bf16* Wb    = (__bf16*)(ws + 71827456);            //    294,912 B

  // 1) softmax(T_param) -> bf16, transposed per symbol: [I][j][i]
  softmax_T_kernel<<<(NS * NI) / 8, 256, 0, stream>>>(Tp, Tsm);
  // 2) pack out_W -> bf16 [O][288]
  pack_w_kernel<<<(NO * KPAD) / 256, 256, 0, stream>>>(W, Wb);
  // 3) sequential scan: TDM double buffering + WMMA matvec (270,912 B LDS)
  hipFuncSetAttribute(reinterpret_cast<const void*>(&scan_kernel),
                      hipFuncAttributeMaxDynamicSharedMemorySize, (int)SMEMB);
  scan_kernel<<<1, 256, SMEMB, stream>>>(seq, Tsm, delta, feats);
  // 4) WMMA output GEMM + softmax
  out_gemm_kernel<<<SEQ_LEN / 16, 256, 0, stream>>>(feats, Wb, bias, out);
}
